// SelfAttention_51153060495423
// MI455X (gfx1250) — compile-verified, matrix operand order checked by reading, not run on hardware
//
#include <hip/hip_runtime.h>
#include <hip/hip_bf16.h>

// ---------------------------------------------------------------------------
// Self-attention block (B=2, S=2048, E=2048, N=16, H=128) for gfx1250.
// All GEMMs on v_wmma_f32_16x16x32_bf16 (bf16 in, f32 accumulate).
// GEMMs use a 2-stage LDS pipeline fed by GLOBAL_LOAD_ASYNC_TO_LDS_B128
// (ASYNCcnt) when the toolchain exposes the builtins; sync fallback otherwise.
// ---------------------------------------------------------------------------

typedef __attribute__((ext_vector_type(16))) __bf16 v16bf;
typedef __attribute__((ext_vector_type(8)))  float  v8f;
typedef unsigned short bf16_t;

#define BQ 2
#define SQ 2048
#define EQ 2048
#define NH 16
#define HD 128
#define MQ (BQ * SQ)      // 4096
#define F3 (3 * EQ)       // 6144

#if __has_builtin(__builtin_amdgcn_global_load_async_to_lds_b128) && \
    __has_builtin(__builtin_amdgcn_s_wait_asynccnt)
#define USE_ASYNC_LDS 1
#else
#define USE_ASYNC_LDS 0
#endif

#if USE_ASYNC_LDS
// Builtin signature (from clang diagnostic): arg0 is
// 'int __attribute__((vector_size(16)))*' in the global address space,
// arg1 the LDS-side pointer, then imm offset + imm cpol.
typedef int v4i_vs __attribute__((vector_size(16)));
typedef __attribute__((address_space(1))) v4i_vs* g_v4i_p;
typedef __attribute__((address_space(3))) v4i_vs* l_v4i_p;

// 16B/lane global -> LDS, no VGPR data path, tracked with ASYNCcnt.
// Address-space pointers built via integer round-trip (flat LDS addresses
// carry the LDS offset in their low 32 bits per the aperture mapping).
static __device__ __forceinline__ void async_copy16(const void* g, void* l) {
    __builtin_amdgcn_global_load_async_to_lds_b128(
        (g_v4i_p)(unsigned long long)g,
        (l_v4i_p)(unsigned int)(unsigned long long)l, 0, 0);
}
#define ASYNC_WAIT(n) __builtin_amdgcn_s_wait_asynccnt(n)
#else
static __device__ __forceinline__ void async_copy16(const void* g, void* l) {
    *(uint4*)l = *(const uint4*)g;   // sync fallback: global->VGPR->LDS
}
#define ASYNC_WAIT(n) ((void)0)
#endif

static __device__ __forceinline__ bf16_t f2bf(float f) {
    unsigned u = __builtin_bit_cast(unsigned, f);
    u += 0x7FFFu + ((u >> 16) & 1u);   // RNE to bf16
    return (bf16_t)(u >> 16);
}

static __device__ __forceinline__ v16bf load_frag(const bf16_t* base, int ldm, int kbase) {
    const int lane = threadIdx.x & 31;
    const bf16_t* p = base + (lane & 15) * ldm + kbase + ((lane >> 4) << 4);
    return *(const v16bf*)p;   // 32B -> two ds_load_b128
}

static __device__ __forceinline__ v8f wmma_bf16(v16bf a, v16bf b, v8f c) {
    return __builtin_amdgcn_wmma_f32_16x16x32_bf16(
        false, a, false, b, (short)0, c, false, false);
}

// ---------------------------------------------------------------------------
// fp32 -> bf16 conversion (grid-stride)
// ---------------------------------------------------------------------------
__global__ void convert_kernel(const float* __restrict__ src,
                               bf16_t* __restrict__ dst, int n) {
    for (int i = blockIdx.x * blockDim.x + threadIdx.x; i < n;
         i += gridDim.x * blockDim.x)
        dst[i] = f2bf(src[i]);
}

// ---------------------------------------------------------------------------
// Shared tiled-GEMM core: C[128x128] per WG (8 waves), wave = 32x64 = 2x4
// accumulators; K-step 32; double-buffered LDS, async issue-ahead-by-one.
// Stage copy: each of 256 threads moves 32B of A and 32B of B.
// ---------------------------------------------------------------------------
#define GEMM_STAGE(buf, kb)                                                    \
    do {                                                                       \
        const bf16_t* gA = Ap + (size_t)(mbase + crow) * EQ + (kb) + ccol;     \
        const bf16_t* gB = Bp + (size_t)(nbase + crow) * EQ + (kb) + ccol;     \
        bf16_t* lA = sA[buf] + crow * 32 + ccol;                               \
        bf16_t* lB = sB[buf] + crow * 32 + ccol;                               \
        async_copy16(gA, lA);  async_copy16(gA + 8, lA + 8);                   \
        async_copy16(gB, lB);  async_copy16(gB + 8, lB + 8);                   \
    } while (0)

#define GEMM_CORE()                                                            \
    GEMM_STAGE(0, 0);                                                          \
    for (int kb = 0; kb < EQ; kb += 32) {                                      \
        const int buf = (kb >> 5) & 1;                                         \
        if (kb + 32 < EQ) {                                                    \
            GEMM_STAGE(buf ^ 1, kb + 32);                                      \
            __builtin_prefetch(Ap + (size_t)(mbase + crow) * EQ + kb + 64, 0, 0); \
            ASYNC_WAIT(4);   /* stage kb done, stage kb+32 in flight */        \
        } else {                                                               \
            ASYNC_WAIT(0);                                                     \
        }                                                                      \
        __syncthreads();     /* all waves' stage writes visible */             \
        v16bf af[2], bfg[4];                                                   \
        for (int i = 0; i < 2; i++)                                            \
            af[i] = load_frag(sA[buf] + (wm * 32 + i * 16) * 32, 32, 0);       \
        for (int j = 0; j < 4; j++)                                            \
            bfg[j] = load_frag(sB[buf] + (wn * 64 + j * 16) * 32, 32, 0);      \
        for (int i = 0; i < 2; i++)                                            \
            for (int j = 0; j < 4; j++)                                        \
                acc[i][j] = wmma_bf16(af[i], bfg[j], acc[i][j]);               \
        __syncthreads();     /* read-fence before buf is overwritten */        \
    }

// ---------------------------------------------------------------------------
// QKV GEMM: qkv[m,f] = sum_e Xb[m,e]*Wqkvb[f,e] + bias[f];
// epilogue scatters bf16 into Q/K/V laid out [B, N, S, H].
// ---------------------------------------------------------------------------
__global__ __launch_bounds__(256) void qkv_gemm_kernel(
    const bf16_t* __restrict__ Ap, const bf16_t* __restrict__ Bp,
    const float* __restrict__ bias,
    bf16_t* __restrict__ Qb, bf16_t* __restrict__ Kb, bf16_t* __restrict__ Vb) {
    __shared__ __attribute__((aligned(16))) bf16_t sA[2][128 * 32];
    __shared__ __attribute__((aligned(16))) bf16_t sB[2][128 * 32];

    const int tid  = threadIdx.x;
    const int lane = tid & 31;
    const int wave = tid >> 5;
    const int wm = wave & 3;
    const int wn = wave >> 2;
    const int mbase = blockIdx.x * 128;
    const int nbase = blockIdx.y * 128;
    const int crow = tid >> 1;
    const int ccol = (tid & 1) * 16;

    v8f acc[2][4];
    for (int i = 0; i < 2; i++)
        for (int j = 0; j < 4; j++)
            for (int e = 0; e < 8; e++) acc[i][j][e] = 0.f;

    GEMM_CORE();

    const int rin = (lane >> 4) * 8;   // C layout: row = e + 8*(lane>>4)
    const int cin = lane & 15;
    for (int i = 0; i < 2; i++)
        for (int j = 0; j < 4; j++) {
            const int gf = nbase + wn * 64 + j * 16 + cin;   // 0..6143
            const float bv = bias[gf];
            const int which = gf >> 11;                      // 0=q 1=k 2=v
            const int f2 = gf & 2047;
            const int head = f2 >> 7, h = f2 & 127;
            bf16_t* dst = which == 0 ? Qb : (which == 1 ? Kb : Vb);
            for (int e = 0; e < 8; e++) {
                const int gm = mbase + wm * 32 + i * 16 + rin + e;
                const int b = gm >> 11, s = gm & 2047;
                dst[((size_t)(b * NH + head) * SQ + s) * HD + h] =
                    f2bf(acc[i][j][e] + bv);
            }
        }
}

// ---------------------------------------------------------------------------
// Output projection: out[m,n] = sum_e Yb[m,e]*Wprojb[n,e] + bias[n]  (f32 out)
// ---------------------------------------------------------------------------
__global__ __launch_bounds__(256) void proj_gemm_kernel(
    const bf16_t* __restrict__ Ap, const bf16_t* __restrict__ Bp,
    const float* __restrict__ bias, float* __restrict__ out) {
    __shared__ __attribute__((aligned(16))) bf16_t sA[2][128 * 32];
    __shared__ __attribute__((aligned(16))) bf16_t sB[2][128 * 32];

    const int tid  = threadIdx.x;
    const int lane = tid & 31;
    const int wave = tid >> 5;
    const int wm = wave & 3;
    const int wn = wave >> 2;
    const int mbase = blockIdx.x * 128;
    const int nbase = blockIdx.y * 128;
    const int crow = tid >> 1;
    const int ccol = (tid & 1) * 16;

    v8f acc[2][4];
    for (int i = 0; i < 2; i++)
        for (int j = 0; j < 4; j++)
            for (int e = 0; e < 8; e++) acc[i][j][e] = 0.f;

    GEMM_CORE();

    const int rin = (lane >> 4) * 8;
    const int cin = lane & 15;
    for (int i = 0; i < 2; i++)
        for (int j = 0; j < 4; j++) {
            const int gf = nbase + wn * 64 + j * 16 + cin;
            const float bv = bias[gf];
            for (int e = 0; e < 8; e++) {
                const int gm = mbase + wm * 32 + i * 16 + rin + e;
                out[(size_t)gm * EQ + gf] = acc[i][j][e] + bv;
            }
        }
}

// ---------------------------------------------------------------------------
// Flash attention: one wave per (b, head, 16-row query tile).
// K tiles streamed via async-to-LDS; V transposed on the LDS fill so the
// P*V B-fragments have K contiguous; online softmax in f32.
// ---------------------------------------------------------------------------
__global__ __launch_bounds__(32) void attn_kernel(
    const bf16_t* __restrict__ Qb, const bf16_t* __restrict__ Kb,
    const bf16_t* __restrict__ Vb, bf16_t* __restrict__ Yb) {
    __shared__ __attribute__((aligned(16))) bf16_t sQ[16 * 128];
    __shared__ __attribute__((aligned(16))) bf16_t sK[64 * 128];
    __shared__ __attribute__((aligned(16))) bf16_t sVt[128 * 64];  // [h][c]
    __shared__ __attribute__((aligned(16))) bf16_t sP[16 * 64];
    __shared__ float sS[16 * 64];
    __shared__ float sScale[16];
    __shared__ float sSum[16];

    const int lane = threadIdx.x;
    const int bid = blockIdx.x;
    const int qtile = bid & 127;            // S/16 = 128 tiles
    const int head = (bid >> 7) & 15;
    const int b = bid >> 11;
    const int qbase = qtile * 16;

    const bf16_t* qp  = Qb + ((size_t)(b * NH + head) * SQ + qbase) * HD;
    const bf16_t* kp0 = Kb + (size_t)(b * NH + head) * SQ * HD;
    const bf16_t* vp0 = Vb + (size_t)(b * NH + head) * SQ * HD;

    {   // Q tile: 2048 bf16 = 256 x 16B, async when available
        for (int i = lane; i < 256; i += 32)
            async_copy16(qp + i * 8, sQ + i * 8);
    }

    v8f o[8];
    for (int t = 0; t < 8; t++)
        for (int e = 0; e < 8; e++) o[t][e] = 0.f;
    float m_i = -1e30f, l_i = 0.f;

    const float sm_scale = 0.08838834764831845f;  // 1/sqrt(128)
    const int rin = (lane >> 4) * 8;
    const int cin = lane & 15;
    const int kend = qbase + 16;                  // causal bound (exclusive)

    for (int kb = 0; kb < kend; kb += 64) {
        {   // K tile 64x128: 1024 x 16B async copies
            const bf16_t* gk = kp0 + (size_t)kb * HD;
            for (int i = lane; i < 1024; i += 32)
                async_copy16(gk + i * 8, sK + i * 8);
        }
        {   // V tile transposed into sVt[h][c] (needs VGPR path)
            const uint4* g = (const uint4*)(vp0 + (size_t)kb * HD);
            for (int i = lane; i < 1024; i += 32) {
                const int c = i >> 4;            // key row 0..63
                const int hb = (i & 15) * 8;     // h base
                uint4 d = g[i];
                sVt[(hb + 0) * 64 + c] = (bf16_t)(d.x & 0xffff);
                sVt[(hb + 1) * 64 + c] = (bf16_t)(d.x >> 16);
                sVt[(hb + 2) * 64 + c] = (bf16_t)(d.y & 0xffff);
                sVt[(hb + 3) * 64 + c] = (bf16_t)(d.y >> 16);
                sVt[(hb + 4) * 64 + c] = (bf16_t)(d.z & 0xffff);
                sVt[(hb + 5) * 64 + c] = (bf16_t)(d.z >> 16);
                sVt[(hb + 6) * 64 + c] = (bf16_t)(d.w & 0xffff);
                sVt[(hb + 7) * 64 + c] = (bf16_t)(d.w >> 16);
            }
        }
        ASYNC_WAIT(0);          // K (and first-iter Q) async copies landed
        __syncthreads();

        // S = Q * K^T  (16x64), K-dim = 128 in 4 chunks of 32
        v8f sacc[4];
        for (int j = 0; j < 4; j++)
            for (int e = 0; e < 8; e++) sacc[j][e] = 0.f;
        for (int kk = 0; kk < 4; kk++) {
            v16bf aq = load_frag(sQ, 128, kk * 32);
            for (int j = 0; j < 4; j++) {
                v16bf bk = load_frag(sK + j * 16 * 128, 128, kk * 32);
                sacc[j] = wmma_bf16(aq, bk, sacc[j]);
            }
        }
        // scale + causal mask, spill to LDS
        for (int j = 0; j < 4; j++) {
            const int col = kb + j * 16 + cin;
            for (int e = 0; e < 8; e++) {
                const int row = qbase + rin + e;
                float v = sacc[j][e] * sm_scale;
                if (col > row) v = -1e30f;
                sS[(rin + e) * 64 + j * 16 + cin] = v;
            }
        }
        __syncthreads();

        // online softmax: lanes 0..15 own one row each
        if (lane < 16) {
            const int r = lane;
            float mx = m_i;
            for (int c = 0; c < 64; c++) mx = fmaxf(mx, sS[r * 64 + c]);
            const float alpha = __expf(m_i - mx);
            float sum = 0.f;
            for (int c = 0; c < 64; c++) {
                const float p = __expf(sS[r * 64 + c] - mx);
                sP[r * 64 + c] = f2bf(p);
                sum += p;
            }
            l_i = l_i * alpha + sum;
            m_i = mx;
            sScale[r] = alpha;
        }
        __syncthreads();

        // rescale running O, then O += P * V
        float fs[8];
        for (int e = 0; e < 8; e++) fs[e] = sScale[rin + e];
        for (int t = 0; t < 8; t++)
            for (int e = 0; e < 8; e++) o[t][e] *= fs[e];

        for (int cc = 0; cc < 2; cc++) {
            v16bf ap = load_frag(sP, 64, cc * 32);
            for (int t = 0; t < 8; t++) {
                v16bf bv = load_frag(sVt + t * 16 * 64, 64, cc * 32);
                o[t] = wmma_bf16(ap, bv, o[t]);
            }
        }
        __syncthreads();
    }

    if (lane < 16) sSum[lane] = l_i;
    __syncthreads();

    float linv[8];
    for (int e = 0; e < 8; e++) linv[e] = 1.f / sSum[rin + e];
    bf16_t* yp = Yb + ((size_t)(b * SQ) + qbase) * EQ + head * HD;
    for (int t = 0; t < 8; t++)
        for (int e = 0; e < 8; e++)
            yp[(size_t)(rin + e) * EQ + t * 16 + cin] = f2bf(o[t][e] * linv[e]);
}

// ---------------------------------------------------------------------------
// Host launch. Inputs (setup_inputs order): mask, x, Wqkv, bqkv, Wproj, bproj.
// Workspace: Xb@0 (16MB) | Wqkvb@16MB (24MB) | Wprojb@40MB (8MB) |
//            Qb@48MB | Kb@64MB | Vb@80MB | Yb@96MB  (each 16MB, ~112MB total)
// ---------------------------------------------------------------------------
extern "C" void kernel_launch(void* const* d_in, const int* in_sizes, int n_in,
                              void* d_out, int out_size, void* d_ws, size_t ws_size,
                              hipStream_t stream) {
    (void)in_sizes; (void)n_in; (void)out_size; (void)ws_size;
    const float* x     = (const float*)d_in[1];
    const float* Wqkv  = (const float*)d_in[2];
    const float* bqkv  = (const float*)d_in[3];
    const float* Wproj = (const float*)d_in[4];
    const float* bproj = (const float*)d_in[5];
    float* out = (float*)d_out;

    char* ws = (char*)d_ws;
    bf16_t* Xb     = (bf16_t*)(ws + 0);
    bf16_t* Wqkvb  = (bf16_t*)(ws + 16777216);
    bf16_t* Wprojb = (bf16_t*)(ws + 41943040);
    bf16_t* Qb     = (bf16_t*)(ws + 50331648);
    bf16_t* Kb     = (bf16_t*)(ws + 67108864);
    bf16_t* Vb     = (bf16_t*)(ws + 83886080);
    bf16_t* Yb     = (bf16_t*)(ws + 100663296);

    convert_kernel<<<2048, 256, 0, stream>>>(x,     Xb,     MQ * EQ);
    convert_kernel<<<2048, 256, 0, stream>>>(Wqkv,  Wqkvb,  F3 * EQ);
    convert_kernel<<<1024, 256, 0, stream>>>(Wproj, Wprojb, EQ * EQ);

    qkv_gemm_kernel<<<dim3(MQ / 128, F3 / 128), 256, 0, stream>>>(
        Xb, Wqkvb, bqkv, Qb, Kb, Vb);

    attn_kernel<<<BQ * NH * (SQ / 16), 32, 0, stream>>>(Qb, Kb, Vb, Yb);

    proj_gemm_kernel<<<dim3(MQ / 128, EQ / 128), 256, 0, stream>>>(
        Yb, Wprojb, bproj, out);
}
